// EGNN_23914377904397
// MI455X (gfx1250) — compile-verified, matrix operand order checked by reading, not run on hardware
//
#include <hip/hip_runtime.h>
#include <hip/hip_bf16.h>

// ---------------------------------------------------------------------------
// EGNN forward for MI455X (gfx1250): all GEMMs via v_wmma_f32_16x16x32_f16.
// M=32 per wave (two 16-row subtiles) so each B fragment feeds 2 WMMAs;
// B fragments batch-loaded per k-step; A fragments gathered straight from
// global (row == lane&15 in the WMMA A layout, so no LDS round trip).
// ---------------------------------------------------------------------------

typedef __attribute__((ext_vector_type(16))) _Float16 v16h;
typedef __attribute__((ext_vector_type(8)))  _Float16 v8h;
typedef __attribute__((ext_vector_type(8)))  float    v8f;

#define HDIM 128
#define TSTRIDE 136   // LDS row stride in halves (272B, bank-conflict friendly)

static __device__ __forceinline__ v8f wmma_f16(v16h a, v16h b, v8f c) {
    return __builtin_amdgcn_wmma_f32_16x16x32_f16(
        /*neg_a=*/false, a, /*neg_b=*/false, b,
        /*c_mod=*/(short)0, c, /*reuse_a=*/false, /*reuse_b=*/false);
}

// A fragment (16x32 f16) per ISA layout: lane<16: M=lane, K=0..7|16..23 ;
// lane>=16: M=lane-16, K=8..15|24..31.  Row is implicit in the base pointer.
static __device__ __forceinline__ v16h load_a_frag(const _Float16* rowbase,
                                                   int kt, int lane) {
    const int kbase = (kt << 5) + ((lane >> 4) << 3);
    const v8h* p0 = (const v8h*)(rowbase + kbase);
    const v8h* p1 = (const v8h*)(rowbase + kbase + 16);
    v8h lo = *p0, hi = *p1;
    return __builtin_shufflevector(lo, hi, 0, 1, 2, 3, 4, 5, 6, 7,
                                   8, 9, 10, 11, 12, 13, 14, 15);
}

// Same fragment but converting from an f32 row (for aggr operand).
static __device__ __forceinline__ v16h load_a_frag_f32(const float* rowbase,
                                                       int kt, int lane) {
    const int kbase = (kt << 5) + ((lane >> 4) << 3);
    const float* p = rowbase + kbase;
    v16h a;
#pragma unroll
    for (int i = 0; i < 8; ++i) a[i] = (_Float16)p[i];
#pragma unroll
    for (int i = 0; i < 8; ++i) a[8 + i] = (_Float16)p[16 + i];
    return a;
}

// B fragments pre-packed so each lane's 16 halves are contiguous (32B load).
static __device__ __forceinline__ v16h load_b_frag(const _Float16* W,
                                                   int kt, int nt, int lane) {
    return *(const v16h*)(W + ((((kt << 3) + nt) << 5) + lane) * 16);
}

// ---------------------------------------------------------------------------
// Weight pack: [128,128] f32 row-major (k,n) -> f16 B-fragment order
// ---------------------------------------------------------------------------
__global__ void egnn_pack_b(const float* __restrict__ src, _Float16* __restrict__ dst) {
    int i = blockIdx.x * blockDim.x + threadIdx.x;
    if (i >= HDIM * HDIM) return;
    int idx = i;
    int ii = idx & 15;  idx >>= 4;
    int lane = idx & 31; idx >>= 5;
    int nt = idx & 7;   idx >>= 3;
    int kt = idx;       // 0..3
    int n = (nt << 4) + (lane & 15);
    int khalf = (lane >> 4) << 3;
    int koff = (ii < 8) ? (khalf + ii) : (16 + khalf + (ii - 8));
    int k = (kt << 5) + koff;
    dst[i] = (_Float16)src[k * HDIM + n];
}

__global__ void egnn_embed(const float* __restrict__ x, const float* __restrict__ w,
                           const float* __restrict__ b, float* __restrict__ h,
                           _Float16* __restrict__ h16, int total) {
    int i = blockIdx.x * blockDim.x + threadIdx.x;
    if (i >= total) return;
    int n = i >> 7, c = i & 127;
    float v = x[n] * w[c] + b[c];
    h[i] = v;
    h16[i] = (_Float16)v;
}

__global__ void egnn_dist(const float* __restrict__ pos,
                          const long long* __restrict__ row,
                          const long long* __restrict__ col,
                          float* __restrict__ dist, int E) {
    int e = blockIdx.x * blockDim.x + threadIdx.x;
    if (e >= E) return;
    int a = (int)row[e], b = (int)col[e];
    float dx = pos[a * 3 + 0] - pos[b * 3 + 0];
    float dy = pos[a * 3 + 1] - pos[b * 3 + 1];
    float dz = pos[a * 3 + 2] - pos[b * 3 + 2];
    dist[e] = dx * dx + dy * dy + dz * dz;
}

__global__ void egnn_zero(float* __restrict__ p, long long n) {
    long long i = (long long)blockIdx.x * blockDim.x + threadIdx.x;
    if (i < n) p[i] = 0.0f;
}

// ---------------------------------------------------------------------------
// Edge message kernel: one wave = 32 edges (two 16-row WMMA subtiles).
//   m = silu(x_i@W1a + x_j@W1b + dist*w1d + b1) @ W2 + b2 ; atomic scatter
// ---------------------------------------------------------------------------
__global__ __launch_bounds__(256) void egnn_edge_kernel(
    const _Float16* __restrict__ h16,
    const long long* __restrict__ rowIdx,
    const long long* __restrict__ colIdx,
    const float* __restrict__ dist,
    const _Float16* __restrict__ W1a,
    const _Float16* __restrict__ W1b,
    const float* __restrict__ w1d,
    const float* __restrict__ b1,
    const _Float16* __restrict__ W2,
    const float* __restrict__ b2,
    float* __restrict__ aggr,
    int nEdges) {
    __shared__ _Float16 tstage[8][32 * TSTRIDE];
    __shared__ float sdist[8][32];
    __shared__ int scol[8][32];

    const int wave = threadIdx.x >> 5;
    const int lane = threadIdx.x & 31;
    const long long ebase = (long long)(blockIdx.x * 8 + wave) * 32;
    if (ebase >= nEdges) return;
    _Float16* T = tstage[wave];

    // stage per-tile scalars (one edge per lane)
    sdist[wave][lane] = dist[ebase + lane];
    scol[wave][lane] = (int)colIdx[ebase + lane];

    const int mr = lane & 15;
    const int ci0 = (int)colIdx[ebase + mr];
    const int ci1 = (int)colIdx[ebase + 16 + mr];
    const int rj0 = (int)rowIdx[ebase + mr];
    const int rj1 = (int)rowIdx[ebase + 16 + mr];
    const _Float16* hci0 = h16 + (size_t)ci0 * HDIM;
    const _Float16* hci1 = h16 + (size_t)ci1 * HDIM;
    const _Float16* hrj0 = h16 + (size_t)rj0 * HDIM;
    const _Float16* hrj1 = h16 + (size_t)rj1 * HDIM;

    v8f acc0[8], acc1[8];
#pragma unroll
    for (int nt = 0; nt < 8; ++nt) {
        v8f z = {0.f, 0.f, 0.f, 0.f, 0.f, 0.f, 0.f, 0.f};
        acc0[nt] = z; acc1[nt] = z;
    }

    // ---- GEMM1a: x_i @ W1a (A gathered straight from global) ----
#pragma unroll
    for (int kt = 0; kt < 4; ++kt) {
        v16h a0 = load_a_frag(hci0, kt, lane);
        v16h a1 = load_a_frag(hci1, kt, lane);
        v16h b[8];
#pragma unroll
        for (int nt = 0; nt < 8; ++nt) b[nt] = load_b_frag(W1a, kt, nt, lane);
#pragma unroll
        for (int nt = 0; nt < 8; ++nt) {
            acc0[nt] = wmma_f16(a0, b[nt], acc0[nt]);
            acc1[nt] = wmma_f16(a1, b[nt], acc1[nt]);
        }
    }
    // ---- GEMM1b: += x_j @ W1b ----
#pragma unroll
    for (int kt = 0; kt < 4; ++kt) {
        v16h a0 = load_a_frag(hrj0, kt, lane);
        v16h a1 = load_a_frag(hrj1, kt, lane);
        v16h b[8];
#pragma unroll
        for (int nt = 0; nt < 8; ++nt) b[nt] = load_b_frag(W1b, kt, nt, lane);
#pragma unroll
        for (int nt = 0; nt < 8; ++nt) {
            acc0[nt] = wmma_f16(a0, b[nt], acc0[nt]);
            acc1[nt] = wmma_f16(a1, b[nt], acc1[nt]);
        }
    }

    // ---- epilogue 1: + dist*w1d + b1, silu, stash f16 tile for GEMM2 ----
    const int mlo = (lane >> 4) << 3;
    const int nc = lane & 15;
#pragma unroll
    for (int nt = 0; nt < 8; ++nt) {
        const int coln = (nt << 4) + nc;
        const float wd = w1d[coln];
        const float bb = b1[coln];
#pragma unroll
        for (int v = 0; v < 8; ++v) {
            const int m = mlo + v;
            float f0 = acc0[nt][v] + sdist[wave][m] * wd + bb;
            float f1 = acc1[nt][v] + sdist[wave][16 + m] * wd + bb;
            float s0 = f0 / (1.0f + __expf(-f0));
            float s1 = f1 / (1.0f + __expf(-f1));
            T[m * TSTRIDE + coln] = (_Float16)s0;
            T[(16 + m) * TSTRIDE + coln] = (_Float16)s1;
            acc0[nt][v] = 0.0f;
            acc1[nt][v] = 0.0f;
        }
    }

    // ---- GEMM2: t @ W2 (A from LDS staging) ----
#pragma unroll
    for (int kt = 0; kt < 4; ++kt) {
        v16h a0 = load_a_frag(T + (size_t)mr * TSTRIDE, kt, lane);
        v16h a1 = load_a_frag(T + (size_t)(16 + mr) * TSTRIDE, kt, lane);
        v16h b[8];
#pragma unroll
        for (int nt = 0; nt < 8; ++nt) b[nt] = load_b_frag(W2, kt, nt, lane);
#pragma unroll
        for (int nt = 0; nt < 8; ++nt) {
            acc0[nt] = wmma_f16(a0, b[nt], acc0[nt]);
            acc1[nt] = wmma_f16(a1, b[nt], acc1[nt]);
        }
    }

    // ---- epilogue 2: + b2, atomic scatter-add into aggr[col] ----
#pragma unroll
    for (int nt = 0; nt < 8; ++nt) {
        const int coln = (nt << 4) + nc;
        const float bb2 = b2[coln];
#pragma unroll
        for (int v = 0; v < 8; ++v) {
            const int m = mlo + v;
            atomicAdd(aggr + (size_t)scol[wave][m] * HDIM + coln, acc0[nt][v] + bb2);
            atomicAdd(aggr + (size_t)scol[wave][16 + m] * HDIM + coln, acc1[nt][v] + bb2);
        }
    }
}

// ---------------------------------------------------------------------------
// Node update kernel: one wave = 32 nodes (two 16-row subtiles).
//   h' = silu(h@V1a + aggr@V1b + b1) @ V2 + b2
// ---------------------------------------------------------------------------
__global__ __launch_bounds__(256) void egnn_node_kernel(
    const _Float16* __restrict__ h16in,
    const float* __restrict__ aggr,
    const _Float16* __restrict__ V1a,
    const _Float16* __restrict__ V1b,
    const float* __restrict__ b1,
    const _Float16* __restrict__ V2,
    const float* __restrict__ b2,
    float* __restrict__ hout,
    _Float16* __restrict__ h16out,
    int nNodes) {
    __shared__ _Float16 tstage[8][32 * TSTRIDE];

    const int wave = threadIdx.x >> 5;
    const int lane = threadIdx.x & 31;
    const int nbase = (blockIdx.x * 8 + wave) * 32;
    if (nbase >= nNodes) return;
    const bool full = (nbase + 32) <= nNodes;     // second subtile valid?
    const int nbase1 = full ? (nbase + 16) : nbase;  // clamp for loads
    _Float16* T = tstage[wave];
    const int mr = lane & 15;

    const _Float16* hr0 = h16in + (size_t)(nbase + mr) * HDIM;
    const _Float16* hr1 = h16in + (size_t)(nbase1 + mr) * HDIM;
    const float* ar0 = aggr + (size_t)(nbase + mr) * HDIM;
    const float* ar1 = aggr + (size_t)(nbase1 + mr) * HDIM;

    v8f acc0[8], acc1[8];
#pragma unroll
    for (int nt = 0; nt < 8; ++nt) {
        v8f z = {0.f, 0.f, 0.f, 0.f, 0.f, 0.f, 0.f, 0.f};
        acc0[nt] = z; acc1[nt] = z;
    }

    // ---- GEMM1a: h @ V1a ----
#pragma unroll
    for (int kt = 0; kt < 4; ++kt) {
        v16h a0 = load_a_frag(hr0, kt, lane);
        v16h a1 = load_a_frag(hr1, kt, lane);
        v16h b[8];
#pragma unroll
        for (int nt = 0; nt < 8; ++nt) b[nt] = load_b_frag(V1a, kt, nt, lane);
#pragma unroll
        for (int nt = 0; nt < 8; ++nt) {
            acc0[nt] = wmma_f16(a0, b[nt], acc0[nt]);
            acc1[nt] = wmma_f16(a1, b[nt], acc1[nt]);
        }
    }
    // ---- GEMM1b: += aggr @ V1b (f32 -> f16 convert in registers) ----
#pragma unroll
    for (int kt = 0; kt < 4; ++kt) {
        v16h a0 = load_a_frag_f32(ar0, kt, lane);
        v16h a1 = load_a_frag_f32(ar1, kt, lane);
        v16h b[8];
#pragma unroll
        for (int nt = 0; nt < 8; ++nt) b[nt] = load_b_frag(V1b, kt, nt, lane);
#pragma unroll
        for (int nt = 0; nt < 8; ++nt) {
            acc0[nt] = wmma_f16(a0, b[nt], acc0[nt]);
            acc1[nt] = wmma_f16(a1, b[nt], acc1[nt]);
        }
    }

    // ---- epilogue 1: silu(+b1) -> LDS f16, reset acc ----
    const int mlo = (lane >> 4) << 3;
    const int nc = lane & 15;
#pragma unroll
    for (int nt = 0; nt < 8; ++nt) {
        const int coln = (nt << 4) + nc;
        const float bb = b1[coln];
#pragma unroll
        for (int v = 0; v < 8; ++v) {
            const int m = mlo + v;
            float f0 = acc0[nt][v] + bb;
            float f1 = acc1[nt][v] + bb;
            float s0 = f0 / (1.0f + __expf(-f0));
            float s1 = f1 / (1.0f + __expf(-f1));
            T[m * TSTRIDE + coln] = (_Float16)s0;
            T[(16 + m) * TSTRIDE + coln] = (_Float16)s1;
            acc0[nt][v] = 0.0f;
            acc1[nt][v] = 0.0f;
        }
    }

    // ---- GEMM2: u @ V2 ----
#pragma unroll
    for (int kt = 0; kt < 4; ++kt) {
        v16h a0 = load_a_frag(T + (size_t)mr * TSTRIDE, kt, lane);
        v16h a1 = load_a_frag(T + (size_t)(16 + mr) * TSTRIDE, kt, lane);
        v16h b[8];
#pragma unroll
        for (int nt = 0; nt < 8; ++nt) b[nt] = load_b_frag(V2, kt, nt, lane);
#pragma unroll
        for (int nt = 0; nt < 8; ++nt) {
            acc0[nt] = wmma_f16(a0, b[nt], acc0[nt]);
            acc1[nt] = wmma_f16(a1, b[nt], acc1[nt]);
        }
    }

    // ---- epilogue 2: + b2, write h (f32) and h16 (f16) ----
#pragma unroll
    for (int nt = 0; nt < 8; ++nt) {
        const int coln = (nt << 4) + nc;
        const float bb2 = b2[coln];
#pragma unroll
        for (int v = 0; v < 8; ++v) {
            const int m = mlo + v;
            float v0 = acc0[nt][v] + bb2;
            size_t gi0 = (size_t)(nbase + m) * HDIM + coln;
            hout[gi0] = v0;
            h16out[gi0] = (_Float16)v0;
            if (full) {
                float v1 = acc1[nt][v] + bb2;
                size_t gi1 = (size_t)(nbase + 16 + m) * HDIM + coln;
                hout[gi1] = v1;
                h16out[gi1] = (_Float16)v1;
            }
        }
    }
}

// ---------------------------------------------------------------------------
// Global add pool + output head
// ---------------------------------------------------------------------------
__global__ void egnn_pool(const float* __restrict__ h, const long long* __restrict__ batch,
                          float* __restrict__ g, int total) {
    int i = blockIdx.x * blockDim.x + threadIdx.x;
    if (i >= total) return;
    int n = i >> 7, c = i & 127;
    atomicAdd(g + (size_t)((int)batch[n]) * HDIM + c, h[i]);
}

__global__ void egnn_out(const float* __restrict__ g, const float* __restrict__ w1,
                         const float* __restrict__ b1, const float* __restrict__ w2,
                         const float* __restrict__ b2, float* __restrict__ out) {
    const int r = blockIdx.x;   // graph
    const int c = threadIdx.x;  // 0..31
    float acc = b1[c];
#pragma unroll 8
    for (int k = 0; k < HDIM; ++k) acc += g[r * HDIM + k] * w1[k * 32 + c];
    float t = acc / (1.0f + __expf(-acc));
    float p = t * w2[c];
#pragma unroll
    for (int off = 16; off > 0; off >>= 1) p += __shfl_down(p, off);
    if (c == 0) out[r] = p + b2[0];
}

// ---------------------------------------------------------------------------
// Host orchestration
// ---------------------------------------------------------------------------
extern "C" void kernel_launch(void* const* d_in, const int* in_sizes, int n_in,
                              void* d_out, int out_size, void* d_ws, size_t ws_size,
                              hipStream_t stream) {
    (void)in_sizes; (void)n_in; (void)out_size; (void)ws_size;
    const int N = 30000, E = 600000, H = 128, L = 4, G = 64;

    const float* x        = (const float*)d_in[0];
    const float* pos      = (const float*)d_in[1];
    const long long* eidx = (const long long*)d_in[2];
    const long long* batch= (const long long*)d_in[3];
    const float* emb_w    = (const float*)d_in[4];
    const float* emb_b    = (const float*)d_in[5];
    const float* msg_w1   = (const float*)d_in[6];
    const float* msg_b1   = (const float*)d_in[7];
    const float* msg_w2   = (const float*)d_in[8];
    const float* msg_b2   = (const float*)d_in[9];
    const float* node_w1  = (const float*)d_in[10];
    const float* node_b1  = (const float*)d_in[11];
    const float* node_w2  = (const float*)d_in[12];
    const float* node_b2  = (const float*)d_in[13];
    const float* out_w1   = (const float*)d_in[14];
    const float* out_b1   = (const float*)d_in[15];
    const float* out_w2   = (const float*)d_in[16];
    const float* out_b2   = (const float*)d_in[17];
    float* out = (float*)d_out;

    char* ws = (char*)d_ws;
    size_t off = 0;
    auto take = [&](size_t bytes) -> void* {
        void* p = ws + off;
        off += (bytes + 255) & ~(size_t)255;
        return p;
    };
    float*     hA   = (float*)take((size_t)N * H * 4);
    float*     hB   = (float*)take((size_t)N * H * 4);
    _Float16*  h16A = (_Float16*)take((size_t)N * H * 2);
    _Float16*  h16B = (_Float16*)take((size_t)N * H * 2);
    float*     aggr = (float*)take((size_t)N * H * 4);
    float*     dist = (float*)take((size_t)E * 4);
    float*     g    = (float*)take((size_t)G * H * 4);
    _Float16*  pW1a = (_Float16*)take((size_t)L * H * H * 2);
    _Float16*  pW1b = (_Float16*)take((size_t)L * H * H * 2);
    _Float16*  pW2m = (_Float16*)take((size_t)L * H * H * 2);
    _Float16*  pV1a = (_Float16*)take((size_t)L * H * H * 2);
    _Float16*  pV1b = (_Float16*)take((size_t)L * H * H * 2);
    _Float16*  pW2n = (_Float16*)take((size_t)L * H * H * 2);

    // ---- pack all weights to f16 WMMA B-fragments ----
    const int pb = (H * H + 255) / 256;
    for (int l = 0; l < L; ++l) {
        egnn_pack_b<<<pb, 256, 0, stream>>>(msg_w1 + (size_t)l * 257 * H,            pW1a + (size_t)l * H * H);
        egnn_pack_b<<<pb, 256, 0, stream>>>(msg_w1 + (size_t)l * 257 * H + 128 * H,  pW1b + (size_t)l * H * H);
        egnn_pack_b<<<pb, 256, 0, stream>>>(msg_w2 + (size_t)l * H * H,              pW2m + (size_t)l * H * H);
        egnn_pack_b<<<pb, 256, 0, stream>>>(node_w1 + (size_t)l * 256 * H,           pV1a + (size_t)l * H * H);
        egnn_pack_b<<<pb, 256, 0, stream>>>(node_w1 + (size_t)l * 256 * H + 128 * H, pV1b + (size_t)l * H * H);
        egnn_pack_b<<<pb, 256, 0, stream>>>(node_w2 + (size_t)l * H * H,             pW2n + (size_t)l * H * H);
    }

    // ---- embed + edge scalar ----
    egnn_embed<<<(N * H + 255) / 256, 256, 0, stream>>>(x, emb_w, emb_b, hA, h16A, N * H);
    egnn_dist<<<(E + 255) / 256, 256, 0, stream>>>(pos, eidx, eidx + E, dist, E);

    float* hc = hA; float* hn = hB;
    _Float16* h16c = h16A; _Float16* h16n = h16B;
    const int edgeTiles = E / 32;                  // 18750 (exact)
    const int nodeTiles = (N + 31) / 32;           // 938 (last tile half)
    const int edgeBlocks = (edgeTiles + 7) / 8;
    const int nodeBlocks = (nodeTiles + 7) / 8;
    const int zeroBlocks = (N * H + 255) / 256;

    for (int l = 0; l < L; ++l) {
        egnn_zero<<<zeroBlocks, 256, 0, stream>>>(aggr, (long long)N * H);
        egnn_edge_kernel<<<edgeBlocks, 256, 0, stream>>>(
            h16c, eidx, eidx + E, dist,
            pW1a + (size_t)l * H * H, pW1b + (size_t)l * H * H,
            msg_w1 + (size_t)l * 257 * H + 256 * H,   // dist row of msg_w1
            msg_b1 + (size_t)l * H,
            pW2m + (size_t)l * H * H, msg_b2 + (size_t)l * H,
            aggr, E);
        egnn_node_kernel<<<nodeBlocks, 256, 0, stream>>>(
            h16c, aggr,
            pV1a + (size_t)l * H * H, pV1b + (size_t)l * H * H,
            node_b1 + (size_t)l * H,
            pW2n + (size_t)l * H * H, node_b2 + (size_t)l * H,
            hn, h16n, N);
        float* tf = hc; hc = hn; hn = tf;
        _Float16* th = h16c; h16c = h16n; h16n = th;
    }

    // ---- pool + output head ----
    egnn_zero<<<(G * H + 255) / 256, 256, 0, stream>>>(g, (long long)G * H);
    egnn_pool<<<(N * H + 255) / 256, 256, 0, stream>>>(hc, batch, g, N * H);
    egnn_out<<<G, 32, 0, stream>>>(g, out_w1, out_b1, out_w2, out_b2, out);
}
